// FallAwareAttention_22651657519401
// MI455X (gfx1250) — matrix operand order repaired and checked by reading, not compiled, not run on hardware
//
#include <hip/hip_runtime.h>
#include <hip/hip_bf16.h>
#include <math.h>

// Problem constants (from reference)
#define B_   32
#define C_   64
#define L_   16384
#define HID_ 16
#define CL_  (C_ * L_)
#define GN_EPS 1e-5f

// Strides (chosen so ALL b128 accesses — global and LDS — are 16B-aligned)
#define XS_STR   136   // x tile row stride (floats); halo: 4 left, 1 right; interior at +4
#define PRE_STR  72    // pre bf16 row stride: 144 B rows -> 16B-aligned fragments
#define GW_STR   72    // gate_w bf16 row stride

typedef __attribute__((ext_vector_type(16))) __bf16 v16bf;
typedef __attribute__((ext_vector_type(8)))  float  v8f;

union FragU { uint4 u4[2]; v16bf v; };

__device__ __forceinline__ unsigned short f2bf(float f) {
    // round-to-nearest-even float -> bf16 bits
    unsigned int u = __float_as_uint(f);
    u += 0x7FFFu + ((u >> 16) & 1u);
    return (unsigned short)(u >> 16);
}

__device__ __forceinline__ float gelu_exact(float v) {
    return 0.5f * v * (1.0f + erff(v * 0.70710678118654752f));
}

// ---------------------------------------------------------------------------
// A-matrix fragment: gate_w[o, c] 16x32 bf16 tile (rows = output channel), LDS.
// ISA layout: lane l<16 -> row M=l, K = {k0+0..7, k0+16..23};
//             lane l+16 -> row M=l, K = {k0+8..15, k0+24..31}.
__device__ __forceinline__ v16bf load_a_frag(const unsigned short* gw,
                                             int ob, int k0, int lane) {
    const int row = ob * 16 + (lane & 15);
    const int kb  = (lane >= 16) ? 8 : 0;
    const unsigned short* base = gw + row * GW_STR + k0 + kb;
    FragU f;
    f.u4[0] = *(const uint4*)(base);        // K = k0+kb    .. +7
    f.u4[1] = *(const uint4*)(base + 16);   // K = k0+kb+16 .. +23
    return f.v;
}

// B-matrix fragment: pre[c, pos] 32x16 bf16 tile from LDS laid out [pos][c].
// ISA layout: lane n<16 -> col N=n, K = {k0+0..7, k0+16..23}; lane n+16 -> col N=n, K = {k0+8..15, k0+24..31}.
__device__ __forceinline__ v16bf load_b_frag(const unsigned short* pre,
                                             int p0, int k0, int lane) {
    const int col = lane & 15;
    const int kh  = (lane >= 16) ? 8 : 0;
    const unsigned short* base = pre + (p0 + col) * PRE_STR + k0 + kh;
    FragU f;
    f.u4[0] = *(const uint4*)(base);
    f.u4[1] = *(const uint4*)(base + 16);
    return f.v;
}

// ---------------------------------------------------------------------------
// Kernel 0: zero accumulators (xsum 2048 + svsum 32 + ysum 32 + y2sum 32)
__global__ __launch_bounds__(256) void faa_init(float* __restrict__ acc) {
    for (int i = threadIdx.x; i < B_ * C_ + 3 * B_; i += 256) acc[i] = 0.0f;
}

// Kernel 0b: gate_w f32 -> bf16 with A-friendly stride (pad zeroed)
__global__ __launch_bounds__(256) void faa_gatecvt(const float* __restrict__ gw,
                                                   unsigned short* __restrict__ gwb) {
    for (int i = threadIdx.x; i < C_ * GW_STR; i += 256) {
        int o = i / GW_STR, c = i - o * GW_STR;
        gwb[i] = (c < C_) ? f2bf(gw[o * C_ + c]) : (unsigned short)0;
    }
}

// ---------------------------------------------------------------------------
// Kernel 1: per-(b,c) sum of x over L, per-b sum of svm over L
__global__ __launch_bounds__(256) void faa_stats(const float* __restrict__ x,
                                                 float* __restrict__ xsum,
                                                 float* __restrict__ svsum) {
    __shared__ float chs[C_];
    __shared__ float svs;
    const int t = threadIdx.x, b = blockIdx.y;
    const int tb = blockIdx.x * 1024;
    const int lane = t & 31;
    if (t < C_) chs[t] = 0.0f;
    if (t == 0) svs = 0.0f;
    __syncthreads();

    const float* xb = x + (size_t)b * CL_;
    float ss0 = 0.f, ss1 = 0.f, ss2 = 0.f, ss3 = 0.f;
    for (int c = 0; c < C_; ++c) {
        const float* xc = xb + c * L_ + tb + t;
        float v0 = xc[0], v1 = xc[256], v2 = xc[512], v3 = xc[768];
        ss0 += v0 * v0; ss1 += v1 * v1; ss2 += v2 * v2; ss3 += v3 * v3;
        float vs = v0 + v1 + v2 + v3;
        #pragma unroll
        for (int off = 16; off > 0; off >>= 1) vs += __shfl_xor(vs, off, 32);
        if (lane == 0) atomicAdd(&chs[c], vs);
    }
    float sv = sqrtf(ss0 + 1e-8f) + sqrtf(ss1 + 1e-8f) +
               sqrtf(ss2 + 1e-8f) + sqrtf(ss3 + 1e-8f);
    #pragma unroll
    for (int off = 16; off > 0; off >>= 1) sv += __shfl_xor(sv, off, 32);
    if (lane == 0) atomicAdd(&svs, sv);
    __syncthreads();

    if (t < C_) atomicAdd(&xsum[b * C_ + t], chs[t]);
    if (t == 0) atomicAdd(&svsum[b], svs);
}

// ---------------------------------------------------------------------------
// Kernel 2: context MLP per batch. svm branch: all channels equal -> hidden
// collapses to b1 + m_sv * rowsum(w1).
__global__ __launch_bounds__(64) void faa_ctx(const float* __restrict__ xsum,
                                              const float* __restrict__ svsum,
                                              const float* __restrict__ w1,
                                              const float* __restrict__ b1,
                                              const float* __restrict__ w2,
                                              const float* __restrict__ b2,
                                              float* __restrict__ ctxo) {
    __shared__ float sm[C_];
    __shared__ float sh[HID_];
    const int b = blockIdx.x, t = threadIdx.x;
    const float invL = 1.0f / (float)L_;
    sm[t] = xsum[b * C_ + t] * invL;
    __syncthreads();
    if (t < HID_) {
        float s = 0.f, ws = 0.f;
        for (int c = 0; c < C_; ++c) { float w = w1[t * C_ + c]; s += w * sm[c]; ws += w; }
        float msv = svsum[b] * invL;
        sh[t] = gelu_exact(b1[t] + s) + gelu_exact(b1[t] + msv * ws);
    }
    __syncthreads();
    float v = 2.0f * b2[t];
    for (int h = 0; h < HID_; ++h) v += w2[t * HID_ + h] * sh[h];
    ctxo[b * C_ + t] = v;
}

// ---------------------------------------------------------------------------
// Kernel 3 (main): fused jerk/dwconv/context -> WMMA gate -> sigmoid -> y,
// with per-batch GroupNorm statistic accumulation.
__global__ __launch_bounds__(256) void faa_main(const float* __restrict__ x,
                                                const float* __restrict__ jw,
                                                const float* __restrict__ jrw,
                                                const float* __restrict__ gate_b,
                                                const float* __restrict__ ctx,
                                                const unsigned short* __restrict__ gwb,
                                                float* __restrict__ ybuf,
                                                float* __restrict__ ysum,
                                                float* __restrict__ y2sum) {
    __shared__ __align__(16) float xs[C_ * XS_STR];               // 34.8 KB
    __shared__ __align__(16) unsigned short preb[128 * PRE_STR];  // 18.0 KB
    __shared__ __align__(16) unsigned short gwl[C_ * GW_STR];     //  9.2 KB
    __shared__ __align__(16) float red[256];

    const int b = blockIdx.y;
    const int tbase = blockIdx.x * 128;
    const int t = threadIdx.x;
    const float* xb = x + (size_t)b * CL_;

    // Stage gate weights (bf16) into LDS once per workgroup: 2304 dwords
    {
        const unsigned int* src = (const unsigned int*)gwb;
        unsigned int* dst = (unsigned int*)gwl;
        #pragma unroll
        for (int i = 0; i < 9; ++i) dst[t + 256 * i] = src[t + 256 * i];
    }

    // Stage x tile, interior vectorized (both sides 16B-aligned):
    // xs[c*XS_STR + 4 + p] = x[b, c, tbase + p], p in [0,128)
    for (int idx = t; idx < C_ * 32; idx += 256) {
        const int c = idx >> 5, m = idx & 31;
        float4 v = *(const float4*)(xb + c * L_ + tbase + 4 * m);
        *(float4*)(&xs[c * XS_STR + 4 + 4 * m]) = v;
    }
    // Halo: indices {0,1,2,3,132} -> global tbase-4 .. tbase-1, tbase+128
    for (int idx = t; idx < C_ * 5; idx += 256) {
        const int c = idx / 5, ii = idx - c * 5;
        const int i = (ii < 4) ? ii : 132;
        const int g = tbase - 4 + i;
        xs[c * XS_STR + i] = (g >= 0 && g < L_) ? xb[c * L_ + g] : 0.0f;
    }
    __syncthreads();

    // pre = dwconv(|jerk|) + dwconv(|jerk_rate|) + context  ->  bf16 [pos][c]
    for (int idx = t; idx < C_ * 128; idx += 256) {
        const int c = idx & 63, pos = idx >> 6;
        const int l = tbase + pos;
        const int base = c * XS_STR + 4 - tbase;   // xs[base + j] = x at global j
        float loc = 0.0f;
        #pragma unroll
        for (int k = 0; k < 3; ++k) {
            int j = l - 1 + k;                     // conv tap position (zero-pad outside)
            if (j >= 0 && j < L_) {
                float jk  = (j >= 1) ? xs[base + j] - xs[base + j - 1] : 0.0f;
                float jkm = (j >= 2) ? xs[base + j - 1] - xs[base + j - 2] : 0.0f;
                float jr  = (j >= 1) ? jk - jkm : 0.0f;
                loc += jw[c * 3 + k] * fabsf(jk) + jrw[c * 3 + k] * fabsf(jr);
            }
        }
        preb[pos * PRE_STR + c] = f2bf(loc + ctx[b * C_ + c]);
    }
    __syncthreads();

    // WMMA gate GEMM: attn[o, pos] = sigmoid(gate_w[o,:] . pre[:, pos] + gate_b[o])
    const int wv = t >> 5, lane = t & 31;
    const int p0 = wv * 16;                        // this wave's 16 positions
    const int pos = p0 + (lane & 15);
    const int gl = tbase + pos;
    float s1 = 0.0f, s2 = 0.0f;

    #pragma unroll
    for (int ob = 0; ob < 4; ++ob) {               // 4 blocks of 16 output channels
        v8f acc = {0.f, 0.f, 0.f, 0.f, 0.f, 0.f, 0.f, 0.f};
        #pragma unroll
        for (int ks = 0; ks < 2; ++ks) {           // K = 64 in two 32-wide steps
            v16bf a  = load_a_frag(gwl, ob, ks * 32, lane);
            v16bf bb = load_b_frag(preb, p0, ks * 32, lane);
            acc = __builtin_amdgcn_wmma_f32_16x16x32_bf16(
                false, a, false, bb, (short)0, acc, false, false);
        }
        // C/D layout: lane n -> N(pos)=n; VGPR r -> M = r (lanes<16) / r+8 (lanes>=16)
        const int obase = ob * 16 + ((lane >= 16) ? 8 : 0);
        #pragma unroll
        for (int r = 0; r < 8; ++r) {
            const int o = obase + r;
            float logit = acc[r] + gate_b[o];
            float attn  = 1.0f / (1.0f + __expf(-logit));
            float yv    = xs[o * XS_STR + pos + 4] * attn;
            ybuf[(size_t)b * CL_ + (size_t)o * L_ + gl] = yv;
            s1 += yv; s2 += yv * yv;
        }
    }

    // Block-reduce GroupNorm stats, one atomic per block
    red[t] = s1; __syncthreads();
    for (int off = 128; off > 0; off >>= 1) { if (t < off) red[t] += red[t + off]; __syncthreads(); }
    if (t == 0) atomicAdd(&ysum[b], red[0]);
    __syncthreads();
    red[t] = s2; __syncthreads();
    for (int off = 128; off > 0; off >>= 1) { if (t < off) red[t] += red[t + off]; __syncthreads(); }
    if (t == 0) atomicAdd(&y2sum[b], red[0]);
}

// ---------------------------------------------------------------------------
// Kernel 4: GroupNorm + residual, float4 vectorized
__global__ __launch_bounds__(256) void faa_final(const float* __restrict__ x,
                                                 const float* __restrict__ ybuf,
                                                 const float* __restrict__ ysum,
                                                 const float* __restrict__ y2sum,
                                                 const float* __restrict__ gnw,
                                                 const float* __restrict__ gnb,
                                                 const float* __restrict__ rsp,
                                                 float* __restrict__ out) {
    const size_t n4 = (size_t)B_ * CL_ / 4;
    size_t i4 = (size_t)blockIdx.x * 256 + threadIdx.x;
    if (i4 >= n4) return;
    const size_t i = i4 * 4;
    const int b = (int)(i / CL_);
    const int c = (int)((i / L_) & (C_ - 1));
    const float inv = 1.0f / (float)CL_;
    const float mu  = ysum[b] * inv;
    const float var = y2sum[b] * inv - mu * mu;
    const float rs  = rsqrtf(var + GN_EPS);
    const float w = gnw[c], bb = gnb[c], r = rsp[0];
    float4 xv = *(const float4*)(x + i);
    float4 yv = *(const float4*)(ybuf + i);
    float4 o;
    o.x = xv.x + r * ((yv.x - mu) * rs * w + bb);
    o.y = xv.y + r * ((yv.y - mu) * rs * w + bb);
    o.z = xv.z + r * ((yv.z - mu) * rs * w + bb);
    o.w = xv.w + r * ((yv.w - mu) * rs * w + bb);
    *(float4*)(out + i) = o;
}

// ---------------------------------------------------------------------------
extern "C" void kernel_launch(void* const* d_in, const int* in_sizes, int n_in,
                              void* d_out, int out_size, void* d_ws, size_t ws_size,
                              hipStream_t stream) {
    (void)in_sizes; (void)n_in; (void)out_size; (void)ws_size;
    const float* x   = (const float*)d_in[0];
    const float* jw  = (const float*)d_in[1];
    const float* jrw = (const float*)d_in[2];
    const float* w1  = (const float*)d_in[3];
    const float* b1  = (const float*)d_in[4];
    const float* w2  = (const float*)d_in[5];
    const float* b2  = (const float*)d_in[6];
    const float* gw  = (const float*)d_in[7];
    const float* gb  = (const float*)d_in[8];
    const float* gnw = (const float*)d_in[9];
    const float* gnb = (const float*)d_in[10];
    const float* rsp = (const float*)d_in[11];
    float* out = (float*)d_out;

    // Workspace layout (floats), y buffer first (16B-aligned throughout)
    float* wsf   = (float*)d_ws;
    float* ybuf  = wsf;                            // B*C*L
    float* xsum  = ybuf + (size_t)B_ * CL_;        // B*C
    float* svsum = xsum + B_ * C_;                 // B
    float* ysum  = svsum + B_;                     // B
    float* y2sum = ysum + B_;                      // B
    float* ctx   = y2sum + B_;                     // B*C
    unsigned short* gwb = (unsigned short*)(ctx + B_ * C_);  // C*GW_STR bf16

    faa_init   <<<1, 256, 0, stream>>>(xsum);
    faa_gatecvt<<<1, 256, 0, stream>>>(gw, gwb);
    faa_stats  <<<dim3(L_ / 1024, B_), 256, 0, stream>>>(x, xsum, svsum);
    faa_ctx    <<<B_, 64, 0, stream>>>(xsum, svsum, w1, b1, w2, b2, ctx);
    faa_main   <<<dim3(L_ / 128, B_), 256, 0, stream>>>(x, jw, jrw, gb, ctx, gwb,
                                                        ybuf, ysum, y2sum);
    const size_t n4 = (size_t)B_ * CL_ / 4;
    faa_final  <<<(unsigned)((n4 + 255) / 256), 256, 0, stream>>>(x, ybuf, ysum, y2sum,
                                                                  gnw, gnb, rsp, out);
}
